// MultiHead_Attention_rel_66494683677358
// MI455X (gfx1250) — matrix operand order repaired and checked by reading, not compiled, not run on hardware
//
#include <hip/hip_runtime.h>

#define BB 8
#define LL 384
#define HIDN 512
#define NHN 8
#define DHN 64
#define PLEN 768   // 2*MAX_SEQ_LEN

typedef __attribute__((ext_vector_type(16))) __bf16 v16bf;
typedef __attribute__((ext_vector_type(8)))  float  v8f;

union Tile16 { uint4 q[2]; v16bf v; };

__device__ __forceinline__ unsigned short f2bf(float f) {
  unsigned int u = __float_as_uint(f);
  unsigned int r = (u + 0x7FFFu + ((u >> 16) & 1u)) >> 16;  // RNE
  return (unsigned short)r;
}
__device__ __forceinline__ float bf2f(unsigned short s) {
  return __uint_as_float(((unsigned int)s) << 16);
}

// ---------------------------------------------------------------- fp32 -> bf16
__global__ void convert_f32_bf16(const float* __restrict__ src,
                                 unsigned short* __restrict__ dst, int n) {
  int i = blockIdx.x * blockDim.x + threadIdx.x;
  for (; i < n; i += gridDim.x * blockDim.x) dst[i] = f2bf(src[i]);
}

// A 16x32 bf16 tile: a[0..15]=K(16*half+0..15)
__device__ __forceinline__ void load_a(Tile16& a, const unsigned short* arow, int k) {
  a.q[0] = *(const uint4*)(arow + k);
  a.q[1] = *(const uint4*)(arow + k + 8);
}
// B 32x16 bf16 tiles: lane column n, b[0..15]=K(16*half+0..15)
__device__ __forceinline__ void load_b(Tile16* b, const unsigned short* const* brow, int k) {
#pragma unroll
  for (int t = 0; t < 4; ++t) {
    b[t].q[0] = *(const uint4*)(brow[t] + k);
    b[t].q[1] = *(const uint4*)(brow[t] + k + 8);
  }
}

// ---------------------------------------------------------------- generic WMMA GEMM
// Out[m,n] = sum_k A[m,k] * B[n,k]  (+ bias[n])
// A element: A[b*cAb + h*cAh + m*lda + k]  (bf16)
// B element: B[b*cBb + h*cBh + n*ldb + k]  (bf16)
// outmode: 0 = fp32 to Of ; 1 = bf16 to Obf ; 2 = bf16 "V-transpose" scatter
// Per wave: 16x64 output tile. K loop unrolled x2 with explicit ping-pong
// buffers (no register copies): loads for the next 32-wide K slice are issued
// before the 4 back-to-back WMMAs of the current slice. Requires K % 64 == 0.
__global__ __launch_bounds__(128)
void wmma_gemm(const unsigned short* __restrict__ A0, long long cAb, long long cAh, int lda,
               const unsigned short* __restrict__ B0, long long cBb, long long cBh, int ldb,
               float* __restrict__ Of, unsigned short* __restrict__ Obf,
               long long cOb, long long cOh, int ldo,
               const float* __restrict__ bias,
               int M, int N, int K, int outmode)
{
  const int z  = blockIdx.z;
  const int bz = z / NHN, hz = z % NHN;
  const unsigned short* A  = A0 + (long long)bz * cAb + (long long)hz * cAh;
  const unsigned short* Bp = B0 + (long long)bz * cBb + (long long)hz * cBh;
  const long long obase = (long long)bz * cOb + (long long)hz * cOh;

  const int lane = threadIdx.x & 31;
  const int wave = threadIdx.x >> 5;
  const int m0 = blockIdx.y * 64 + wave * 16;
  if (m0 >= M) return;                       // wave-uniform: EXEC stays all-ones
  const int n0   = blockIdx.x * 64;
  const int half = lane >> 4;                // 0: lanes 0-15, 1: lanes 16-31
  const int lm   = lane & 15;

  // Per-lane row pointers (A: row m0+lm; B: column n0+16t+lm), K-origin folded in.
  const unsigned short* arow = A + (long long)(m0 + lm) * lda + 16 * half;
  const unsigned short* brow[4];
#pragma unroll
  for (int t = 0; t < 4; ++t)
    brow[t] = Bp + (long long)(n0 + 16 * t + lm) * ldb + 16 * half;

  v8f zero = {0.f,0.f,0.f,0.f,0.f,0.f,0.f,0.f};
  v8f acc[4] = {zero, zero, zero, zero};

  // ---- ping-pong buffers; prologue loads slice k=0 into buffer 0
  Tile16 a0, b0[4], a1, b1[4];
  load_a(a0, arow, 0);
  load_b(b0, brow, 0);

  for (int k0 = 0; k0 < K; k0 += 64) {
    // slice k0+32 into buffer 1 (always in range: K % 64 == 0)
    load_a(a1, arow, k0 + 32);
    load_b(b1, brow, k0 + 32);
#pragma unroll
    for (int t = 0; t < 4; ++t)
      acc[t] = __builtin_amdgcn_wmma_f32_16x16x32_bf16(
          false, a0.v, false, b0[t].v, (short)0, acc[t], false, false);

    if (k0 + 64 < K) {                       // slice k0+64 into buffer 0
      load_a(a0, arow, k0 + 64);
      load_b(b0, brow, k0 + 64);
    }
#pragma unroll
    for (int t = 0; t < 4; ++t)
      acc[t] = __builtin_amdgcn_wmma_f32_16x16x32_bf16(
          false, a1.v, false, b1[t].v, (short)0, acc[t], false, false);
  }

#pragma unroll
  for (int t = 0; t < 4; ++t) {
    const int n  = n0 + 16 * t + lm;
    const float bs = bias ? bias[n] : 0.0f;
#pragma unroll
    for (int i = 0; i < 8; ++i) {
      const int m = m0 + i + 8 * half;       // C/D layout: VGPR i -> M = i + 8*half
      const float val = acc[t][i] + bs;
      if (outmode == 0) {
        Of[obase + (long long)m * ldo + n] = val;
      } else if (outmode == 1) {
        Obf[obase + (long long)m * ldo + n] = f2bf(val);
      } else {
        // V projection written transposed: [b, h, d, j]
        const int bb = m / LL, jl = m % LL;
        const int hh = n >> 6, d = n & 63;
        Obf[(((long long)bb * NHN + hh) * DHN + d) * LL + jl] = f2bf(val);
      }
    }
  }
}

// ---------------------------------------------------------------- bias dot tables
// Cb[b,h,j] = sum_d u_bias[h,d] * k[b,h,j,d]
__global__ void compute_Cb(const unsigned short* __restrict__ kbf,
                           const float* __restrict__ u_bias, float* __restrict__ Cb) {
  int idx = blockIdx.x * blockDim.x + threadIdx.x;
  if (idx >= BB * NHN * LL) return;
  int j = idx % LL, h = (idx / LL) % NHN, b = idx / (LL * NHN);
  const unsigned short* kr = kbf + ((long long)(b * LL + j)) * HIDN + h * DHN;
  float s = 0.f;
  for (int d = 0; d < DHN; ++d) s += u_bias[h * DHN + d] * bf2f(kr[d]);
  Cb[idx] = s;
}

// Dp[h,p] = sum_d rproj[p,h,d] * v_bias[h,d]
__global__ void compute_Dp(const unsigned short* __restrict__ rproj,
                           const float* __restrict__ v_bias, float* __restrict__ Dp) {
  int idx = blockIdx.x * blockDim.x + threadIdx.x;
  if (idx >= NHN * PLEN) return;
  int p = idx % PLEN, h = idx / PLEN;
  const unsigned short* rr = rproj + (long long)p * HIDN + h * DHN;
  float s = 0.f;
  for (int d = 0; d < DHN; ++d) s += bf2f(rr[d]) * v_bias[h * DHN + d];
  Dp[idx] = s;
}

// ---------------------------------------------------------------- shift + mask + softmax
// score[j] = (S1[row,j] + QR[row, j-i+384] + Cb[b,h,j] + Dp[h, j-i+384]) / 8
// mask j >= seq_len[b] -> -1e15 ; softmax over j ; write bf16 attn row.
__global__ __launch_bounds__(128)
void softmax_attn(const float* __restrict__ S1, const float* __restrict__ QR,
                  const float* __restrict__ Cb, const float* __restrict__ Dp,
                  const int* __restrict__ seq_len, unsigned short* __restrict__ attn)
{
  __shared__ float red[128];
  const int row = blockIdx.x;                  // ((b*NH + h)*L + i)
  const int i = row % LL;
  const int h = (row / LL) % NHN;
  const int b = row / (LL * NHN);
  const int tid = threadIdx.x;
  const int sl = seq_len[b];

  float x[3];
  float mx = -3.0e38f;
#pragma unroll
  for (int t = 0; t < 3; ++t) {
    const int j = tid + 128 * t;
    const int p = j - i + LL;                  // in [1,767]
    float v = S1[(long long)row * LL + j] + QR[(long long)row * PLEN + p]
            + Cb[((long long)b * NHN + h) * LL + j] + Dp[h * PLEN + p];
    v *= 0.125f;                               // 1/sqrt(64)
    if (j >= sl) v = -1.0e15f;
    x[t] = v;
    mx = fmaxf(mx, v);
  }
  red[tid] = mx; __syncthreads();
  for (int s = 64; s > 0; s >>= 1) { if (tid < s) red[tid] = fmaxf(red[tid], red[tid + s]); __syncthreads(); }
  mx = red[0]; __syncthreads();

  float sum = 0.f;
#pragma unroll
  for (int t = 0; t < 3; ++t) { x[t] = __expf(x[t] - mx); sum += x[t]; }
  red[tid] = sum; __syncthreads();
  for (int s = 64; s > 0; s >>= 1) { if (tid < s) red[tid] += red[tid + s]; __syncthreads(); }
  const float inv = 1.0f / red[0];
#pragma unroll
  for (int t = 0; t < 3; ++t) attn[(long long)row * LL + tid + 128 * t] = f2bf(x[t] * inv);
}

// ---------------------------------------------------------------- launcher
extern "C" void kernel_launch(void* const* d_in, const int* in_sizes, int n_in,
                              void* d_out, int out_size, void* d_ws, size_t ws_size,
                              hipStream_t stream)
{
  (void)in_sizes; (void)n_in; (void)out_size; (void)ws_size;

  const float* key    = (const float*)d_in[0];
  const float* query  = (const float*)d_in[1];
  const float* value  = (const float*)d_in[2];
  const int*   seqlen = (const int*)  d_in[3];
  const float* pe     = (const float*)d_in[4];
  const float* Wk     = (const float*)d_in[5];
  const float* bk     = (const float*)d_in[6];
  const float* Wq     = (const float*)d_in[7];
  const float* bq     = (const float*)d_in[8];
  const float* Wv     = (const float*)d_in[9];
  const float* bv     = (const float*)d_in[10];
  const float* Wr     = (const float*)d_in[11];
  const float* br     = (const float*)d_in[12];
  const float* u_bias = (const float*)d_in[13];
  const float* v_bias = (const float*)d_in[14];
  const float* Wf     = (const float*)d_in[15];
  const float* bff    = (const float*)d_in[16];
  float* out = (float*)d_out;

  const int MROWS = BB * LL;              // 3072
  const int NKEY  = MROWS * HIDN;         // 1572864
  const int NW    = HIDN * HIDN;          // 262144
  const int NPE   = PLEN * HIDN;          // 393216

  char* ws = (char*)d_ws;
  size_t off = 0;
  auto take = [&](size_t bytes) -> char* {
    char* p = ws + off;
    off = (off + bytes + 255) & ~(size_t)255;
    return p;
  };

  unsigned short* key_bf  = (unsigned short*)take((size_t)NKEY * 2);
  unsigned short* qry_bf  = (unsigned short*)take((size_t)NKEY * 2);
  unsigned short* val_bf  = (unsigned short*)take((size_t)NKEY * 2);
  unsigned short* pe_bf   = (unsigned short*)take((size_t)NPE * 2);
  unsigned short* Wk_bf   = (unsigned short*)take((size_t)NW * 2);
  unsigned short* Wq_bf   = (unsigned short*)take((size_t)NW * 2);
  unsigned short* Wv_bf   = (unsigned short*)take((size_t)NW * 2);
  unsigned short* Wr_bf   = (unsigned short*)take((size_t)NW * 2);
  unsigned short* Wf_bf   = (unsigned short*)take((size_t)NW * 2);
  unsigned short* q_bf    = (unsigned short*)take((size_t)NKEY * 2);
  unsigned short* k_bf    = (unsigned short*)take((size_t)NKEY * 2);
  unsigned short* vT_bf   = (unsigned short*)take((size_t)NKEY * 2);  // [b,h,d,j]
  unsigned short* rpj_bf  = (unsigned short*)take((size_t)NPE * 2);   // [p, h*64+d]
  float*          S1      = (float*)take((size_t)BB * NHN * LL * LL * 4);
  float*          QR      = (float*)take((size_t)BB * NHN * LL * PLEN * 4);
  float*          Cb      = (float*)take((size_t)BB * NHN * LL * 4);
  float*          Dp      = (float*)take((size_t)NHN * PLEN * 4);
  unsigned short* attn_bf = (unsigned short*)take((size_t)BB * NHN * LL * LL * 2);
  unsigned short* oat_bf  = (unsigned short*)take((size_t)NKEY * 2);  // [b, i, h*64+d]

  auto cg = [](int n) { return dim3((unsigned)((n + 255) / 256)); };

  // ---- fp32 -> bf16 conversions
  convert_f32_bf16<<<cg(NKEY), 256, 0, stream>>>(key,   key_bf, NKEY);
  convert_f32_bf16<<<cg(NKEY), 256, 0, stream>>>(query, qry_bf, NKEY);
  convert_f32_bf16<<<cg(NKEY), 256, 0, stream>>>(value, val_bf, NKEY);
  convert_f32_bf16<<<cg(NPE),  256, 0, stream>>>(pe,    pe_bf,  NPE);
  convert_f32_bf16<<<cg(NW),   256, 0, stream>>>(Wk, Wk_bf, NW);
  convert_f32_bf16<<<cg(NW),   256, 0, stream>>>(Wq, Wq_bf, NW);
  convert_f32_bf16<<<cg(NW),   256, 0, stream>>>(Wv, Wv_bf, NW);
  convert_f32_bf16<<<cg(NW),   256, 0, stream>>>(Wr, Wr_bf, NW);
  convert_f32_bf16<<<cg(NW),   256, 0, stream>>>(Wf, Wf_bf, NW);

  // ---- projections: X @ W.T + b  (M=3072, N=512, K=512)
  wmma_gemm<<<dim3(8, 48, 1), 128, 0, stream>>>(
      key_bf, 0, 0, HIDN,  Wk_bf, 0, 0, HIDN,
      nullptr, k_bf, 0, 0, HIDN, bk, MROWS, HIDN, HIDN, 1);
  wmma_gemm<<<dim3(8, 48, 1), 128, 0, stream>>>(
      qry_bf, 0, 0, HIDN,  Wq_bf, 0, 0, HIDN,
      nullptr, q_bf, 0, 0, HIDN, bq, MROWS, HIDN, HIDN, 1);
  wmma_gemm<<<dim3(8, 48, 1), 128, 0, stream>>>(
      val_bf, 0, 0, HIDN,  Wv_bf, 0, 0, HIDN,
      nullptr, vT_bf, 0, 0, HIDN, bv, MROWS, HIDN, HIDN, 2);  // V transposed
  // rproj[p] = pe[p] @ Wr.T + br  (M=768) — collapses the [L,L,HID] rpe gather
  wmma_gemm<<<dim3(8, 12, 1), 128, 0, stream>>>(
      pe_bf, 0, 0, HIDN,  Wr_bf, 0, 0, HIDN,
      nullptr, rpj_bf, 0, 0, HIDN, br, PLEN, HIDN, HIDN, 1);

  // ---- bias dot tables
  compute_Cb<<<cg(BB * NHN * LL), 256, 0, stream>>>(k_bf, u_bias, Cb);
  compute_Dp<<<cg(NHN * PLEN),    256, 0, stream>>>(rpj_bf, v_bias, Dp);

  // ---- per-head QK^T: S1[b,h,i,j]  (M=N=384, K=64, batch=64)
  wmma_gemm<<<dim3(6, 6, 64), 128, 0, stream>>>(
      q_bf, (long long)LL * HIDN, DHN, HIDN,
      k_bf, (long long)LL * HIDN, DHN, HIDN,
      S1, nullptr, (long long)NHN * LL * LL, (long long)LL * LL, LL,
      nullptr, LL, LL, DHN, 0);

  // ---- per-head QR: QR[b,h,i,p]  (M=384, N=768, K=64, batch=64)
  wmma_gemm<<<dim3(12, 6, 64), 128, 0, stream>>>(
      q_bf, (long long)LL * HIDN, DHN, HIDN,
      rpj_bf, 0, DHN, HIDN,
      QR, nullptr, (long long)NHN * LL * PLEN, (long long)LL * PLEN, PLEN,
      nullptr, LL, PLEN, DHN, 0);

  // ---- rel-shift + bias + mask + softmax -> attn (bf16)
  softmax_attn<<<dim3(BB * NHN * LL), 128, 0, stream>>>(S1, QR, Cb, Dp, seqlen, attn_bf);

  // ---- attn @ V  (M=384, N=64, K=384, batch=64) -> [b, i, h*64+d] bf16
  wmma_gemm<<<dim3(1, 6, 64), 128, 0, stream>>>(
      attn_bf, (long long)NHN * LL * LL, (long long)LL * LL, LL,
      vT_bf, (long long)NHN * DHN * LL, (long long)DHN * LL, LL,
      nullptr, oat_bf, (long long)LL * HIDN, DHN, HIDN,
      nullptr, LL, DHN, LL, 1);

  // ---- final projection: out = oat @ Wf.T + bf  (fp32 to d_out)
  wmma_gemm<<<dim3(8, 48, 1), 128, 0, stream>>>(
      oat_bf, 0, 0, HIDN,  Wf_bf, 0, 0, HIDN,
      out, nullptr, 0, 0, HIDN, bff, MROWS, HIDN, HIDN, 0);
}
